// SimilarityLossLayer_5385888989384
// MI455X (gfx1250) — matrix operand order repaired and checked by reading, not compiled, not run on hardware
//
#include <hip/hip_runtime.h>

typedef __attribute__((ext_vector_type(16))) __bf16 v16bf;  // 8 VGPRs
typedef __attribute__((ext_vector_type(8)))  __bf16 v8bf;   // 16-byte chunk
typedef __attribute__((ext_vector_type(8)))  float  v8f;

#define NN 8192
#define DD 256
#define JC 4
#define BSTRIDE (DD + 8)   // bf16 stride: 528B -> bank step 4; b128 reads hit each bank exactly 2x

static __device__ __forceinline__ unsigned short f2bf_rne(float x) {
  unsigned u = __float_as_uint(x);
  unsigned r = u + 0x7FFFu + ((u >> 16) & 1u);   // round-to-nearest-even
  return (unsigned short)(r >> 16);
}
static __device__ __forceinline__ float bf2f(unsigned short h) {
  return __uint_as_float(((unsigned)h) << 16);
}

// ---------------- kernel 1a: sq[n] = ||samples[n]||^2 (exact fp32), idsf[n] = input1[n,0,7]
__global__ __launch_bounds__(256) void precompute_kernel(
    const float* __restrict__ samples, const float* __restrict__ input1,
    float* __restrict__ sq, float* __restrict__ idsf) {
  int wave = threadIdx.x >> 5;
  int lane = threadIdx.x & 31;
  int row  = blockIdx.x * 8 + wave;
  const float* rp = samples + (size_t)row * DD;
  float s = 0.f;
#pragma unroll
  for (int q = 0; q < DD / 32; ++q) {
    float v = rp[lane + 32 * q];
    s += v * v;
  }
#pragma unroll
  for (int ofs = 16; ofs > 0; ofs >>= 1) s += __shfl_xor(s, ofs, 32);
  if (lane == 0) {
    sq[row]   = s;
    idsf[row] = input1[(size_t)row * 32 + 7];
  }
}

// ---------------- kernel 1b: compensated bf16 split of samples (hi + lo)
__global__ __launch_bounds__(256) void split_kernel(
    const float* __restrict__ samples,
    unsigned short* __restrict__ hi, unsigned short* __restrict__ lo) {
  size_t i = (size_t)blockIdx.x * 256 + threadIdx.x;
  float x = samples[i];
  unsigned short h = f2bf_rne(x);
  hi[i] = h;
  lo[i] = f2bf_rne(x - bf2f(h));
}

// ---------------- kernel 2: bf16x3 WMMA gram + fused loss, double-buffered staging
__global__ __launch_bounds__(256) void simloss_main_kernel(
    const __bf16* __restrict__ shi, const __bf16* __restrict__ slo,
    const float* __restrict__ sq, const float* __restrict__ idsf,
    float* __restrict__ partial) {
  __shared__ __bf16 Bs_hi[2][16 * BSTRIDE];
  __shared__ __bf16 Bs_lo[2][16 * BSTRIDE];

  const int w    = threadIdx.x >> 5;
  const int lane = threadIdx.x & 31;
  const int half = lane >> 4;            // K-half per ISA 16-bit layouts
  const int l    = lane & 15;            // M row (A) / N col (B, C)

  const int ibase = blockIdx.x * 128 + w * 16;
  const int arow  = ibase + l;

  // A panel 16x256 in registers, both precisions, WMMA 16-bit A layout:
  // group g covers K=[32g..32g+31]; lane half h holds K [32g+8h..+7] and [32g+16+8h..+7].
  v16bf a_hi[DD / 32], a_lo[DD / 32];
  {
    const __bf16* ah = shi + (size_t)arow * DD + 8 * half;
    const __bf16* al = slo + (size_t)arow * DD + 8 * half;
#pragma unroll
    for (int g = 0; g < DD / 32; ++g) {
      v16bf fh, fl;
      ((v8bf*)&fh)[0] = *(const v8bf*)(ah + 32 * g);
      ((v8bf*)&fh)[1] = *(const v8bf*)(ah + 32 * g + 16);
      ((v8bf*)&fl)[0] = *(const v8bf*)(al + 32 * g);
      ((v8bf*)&fl)[1] = *(const v8bf*)(al + 32 * g + 16);
      a_hi[g] = fh;
      a_lo[g] = fl;
    }
  }

  // per-lane row constants: C vgpr r -> row M = r + 8*half
  float sqi[8], idi[8];
#pragma unroll
  for (int r = 0; r < 8; ++r) {
    int m  = ibase + r + 8 * half;
    sqi[r] = sq[m];
    idi[r] = idsf[m];
  }

  float acc[8];
#pragma unroll
  for (int r = 0; r < 8; ++r) acc[r] = 0.f;

  const int jspan = NN / JC;
  const int j0    = blockIdx.y * jspan;
  const int j1    = j0 + jspan;
  const int trow  = threadIdx.x >> 4;    // staging: thread -> (row, 32B chunk)
  const int tcg   = threadIdx.x & 15;

  const size_t soff = (size_t)trow * DD + tcg * 16;   // element offset within a tile row-panel
  const int    doff = trow * BSTRIDE + tcg * 16;      // LDS element offset

  // ---- software pipeline: tile t computes, tile t+1 sits in regs, tile t+2 in flight
  v8bf nh0, nh1, nl0, nl1;
  {
    // tile 0 -> Bs[0]
    const v8bf* sh = (const v8bf*)(shi + (size_t)j0 * DD + soff);
    const v8bf* sl = (const v8bf*)(slo + (size_t)j0 * DD + soff);
    v8bf* dh = (v8bf*)&Bs_hi[0][doff];
    v8bf* dl = (v8bf*)&Bs_lo[0][doff];
    dh[0] = sh[0]; dh[1] = sh[1];
    dl[0] = sl[0]; dl[1] = sl[1];
    // prefetch tile 1 into regs
    int jn = (j0 + 16 < j1) ? j0 + 16 : j0;
    const v8bf* ph = (const v8bf*)(shi + (size_t)jn * DD + soff);
    const v8bf* pl = (const v8bf*)(slo + (size_t)jn * DD + soff);
    nh0 = ph[0]; nh1 = ph[1];
    nl0 = pl[0]; nl1 = pl[1];
  }
  __syncthreads();

  int buf = 0;
  for (int jt = j0; jt < j1; jt += 16) {
    // (1) park tile t+1 into the idle buffer (its loads were issued last iteration)
    {
      v8bf* dh = (v8bf*)&Bs_hi[buf ^ 1][doff];
      v8bf* dl = (v8bf*)&Bs_lo[buf ^ 1][doff];
      dh[0] = nh0; dh[1] = nh1;
      dl[0] = nl0; dl[1] = nl1;
    }
    // (2) launch loads for tile t+2 (in flight during the WMMA block below)
    {
      int j2 = jt + 32;
      int jp = (j2 < j1) ? j2 : j0;          // clamp tail to a valid address
      const v8bf* ph = (const v8bf*)(shi + (size_t)jp * DD + soff);
      const v8bf* pl = (const v8bf*)(slo + (size_t)jp * DD + soff);
      nh0 = ph[0]; nh1 = ph[1];
      nl0 = pl[0]; nl1 = pl[1];
    }

    float sqj = sq[jt + l];
    float idj = idsf[jt + l];

    // (3) compute tile t: three independent accumulator chains
    v8f c_hh = {}, c_hl = {}, c_lh = {};
    const __bf16* bh = &Bs_hi[buf][l * BSTRIDE + 8 * half];
    const __bf16* bl = &Bs_lo[buf][l * BSTRIDE + 8 * half];
#pragma unroll
    for (int g = 0; g < DD / 32; ++g) {
      v16bf fbh, fbl;  // K [32g+8h..+7] and [32g+16+8h..+7] in column n=l
      ((v8bf*)&fbh)[0] = *(const v8bf*)(bh + 32 * g);
      ((v8bf*)&fbh)[1] = *(const v8bf*)(bh + 32 * g + 16);
      ((v8bf*)&fbl)[0] = *(const v8bf*)(bl + 32 * g);
      ((v8bf*)&fbl)[1] = *(const v8bf*)(bl + 32 * g + 16);
      c_hh = __builtin_amdgcn_wmma_f32_16x16x32_bf16(false, a_hi[g], false, fbh,
                                                     (short)0, c_hh, false, false);
      c_hl = __builtin_amdgcn_wmma_f32_16x16x32_bf16(false, a_hi[g], false, fbl,
                                                     (short)0, c_hl, false, false);
      c_lh = __builtin_amdgcn_wmma_f32_16x16x32_bf16(false, a_lo[g], false, fbh,
                                                     (short)0, c_lh, false, false);
    }

    // fused epilogue: (S_ij - [id_i != id_j])^2 accumulated over j
#pragma unroll
    for (int r = 0; r < 8; ++r) {
      float gram = c_hh[r] + c_hl[r] + c_lh[r];
      float S = (sqi[r] + sqj - 2.f * gram) * (1.f / DD);
      float o = (idi[r] != idj) ? 1.f : 0.f;
      float e = S - o;
      acc[r] += e * e;
    }

    // (4) single barrier: publishes Bs[buf^1] for tile t+1 and retires reads of Bs[buf]
    __syncthreads();
    buf ^= 1;
  }

  // collapse N: xor offsets 1..8 stay within each 16-lane half
#pragma unroll
  for (int r = 0; r < 8; ++r) {
    float v = acc[r];
#pragma unroll
    for (int ofs = 1; ofs < 16; ofs <<= 1) v += __shfl_xor(v, ofs, 32);
    acc[r] = v;
  }

  if (l == 0) {
#pragma unroll
    for (int r = 0; r < 8; ++r) {
      int m = ibase + r + 8 * half;
      partial[(size_t)blockIdx.y * NN + m] = acc[r];
    }
  }
}

// ---------------- kernel 3: combine J-chunk partials, scale by 1/N
__global__ __launch_bounds__(256) void finalize_kernel(
    const float* __restrict__ partial, float* __restrict__ out) {
  int i = blockIdx.x * 256 + threadIdx.x;
  float s = 0.f;
#pragma unroll
  for (int jc = 0; jc < JC; ++jc) s += partial[(size_t)jc * NN + i];
  out[i] = s * (1.f / NN);
}

extern "C" void kernel_launch(void* const* d_in, const int* in_sizes, int n_in,
                              void* d_out, int out_size, void* d_ws, size_t ws_size,
                              hipStream_t stream) {
  const float* samples = (const float*)d_in[0];
  const float* input1  = (const float*)d_in[1];

  char* ws = (char*)d_ws;
  float* sq      = (float*)(ws);                         //  32 KB
  float* idsf    = (float*)(ws + 32 * 1024);             //  32 KB
  float* partial = (float*)(ws + 64 * 1024);             // 128 KB (JC * N floats)
  unsigned short* shi = (unsigned short*)(ws + 192 * 1024);                        // 4 MB
  unsigned short* slo = (unsigned short*)(ws + 192 * 1024 + (size_t)NN * DD * 2);  // 4 MB
  float* out = (float*)d_out;

  precompute_kernel<<<NN / 8, 256, 0, stream>>>(samples, input1, sq, idsf);
  split_kernel<<<(NN * DD) / 256, 256, 0, stream>>>(samples, shi, slo);
  simloss_main_kernel<<<dim3(NN / 128, JC), 256, 0, stream>>>(
      (const __bf16*)shi, (const __bf16*)slo, sq, idsf, partial);
  finalize_kernel<<<NN / 256, 256, 0, stream>>>(partial, out);
}
